// DecoderBlock_Mamba_72335839199750
// MI455X (gfx1250) — matrix operand order repaired and checked by reading, not compiled, not run on hardware
//
#include <hip/hip_runtime.h>
#include <hip/hip_bf16.h>
#include <math.h>

typedef __attribute__((ext_vector_type(16))) _Float16 v16h;
typedef __attribute__((ext_vector_type(8)))  _Float16 v8h;
typedef __attribute__((ext_vector_type(8)))  float    v8f;

#define BATCH 4
#define CDIM  64
#define HDIM  64
#define WDIM  64
#define LSEQ  4096          // H*W
#define BL    16384         // BATCH*LSEQ
#define DIN   128
#define DST   16
#define DTR   4
#define NPROJ 36            // DT_RANK + 2*D_STATE
#define NPROJ_PAD 48        // padded to 3 x 16 tiles

// ---------------------------------------------------------------------------
// Multi-tile wave-level WMMA f16 GEMM: one wave computes NT adjacent 16x16
// f32 tiles of C = A(MxK, row-major f16) * W^T, W = [Npad][K] row-major f16
// (rows >= N are zero-padded at conversion time, so all loads are valid and
// the inner loop has no divergence).  K is compile-time -> fully unrolled,
// A fragment loaded once per K-step and reused for NT back-to-back WMMAs.
// ---------------------------------------------------------------------------
template<int K, int NT>
__device__ __forceinline__ void wmma_gemm_multi(const _Float16* __restrict__ A, int lda,
                                                const _Float16* __restrict__ W, int ldb,
                                                int tileM, int tileNbase, v8f (&acc)[NT])
{
    const int lane = threadIdx.x & 31;
    const int half = lane >> 4;     // 0 or 1
    const int idx  = lane & 15;
    const int arow = tileM * 16 + idx;

#pragma unroll
    for (int t = 0; t < NT; ++t) acc[t] = {};

#pragma unroll
    for (int k0 = 0; k0 < K; k0 += 32) {
        // A 16x32 fragment: lane<16 holds row=idx, K={k0..k0+7, k0+16..k0+23};
        // lane>=16 holds the K ranges shifted by 8.  Two b128 loads.
        const _Float16* ap = A + (size_t)arow * lda + k0 + 8 * half;
        v8h alo = *reinterpret_cast<const v8h*>(ap);
        v8h ahi = *reinterpret_cast<const v8h*>(ap + 16);
        v16h a;
#pragma unroll
        for (int i = 0; i < 8; ++i) { a[i] = alo[i]; a[8 + i] = ahi[i]; }

#pragma unroll
        for (int t = 0; t < NT; ++t) {
            // B 32x16 fragment: same K layout as A (K={k0+8h..+7, k0+16+8h..+7}),
            // col = tile col idx.  Two b128 loads, always in-bounds (padded).
            const int bcol = (tileNbase + t) * 16 + idx;
            const _Float16* bp = W + (size_t)bcol * ldb + k0 + 8 * half;
            v8h blo = *reinterpret_cast<const v8h*>(bp);
            v8h bhi = *reinterpret_cast<const v8h*>(bp + 16);
            v16h b;
#pragma unroll
            for (int i = 0; i < 8; ++i) { b[i] = blo[i]; b[8 + i] = bhi[i]; }
            acc[t] = __builtin_amdgcn_wmma_f32_16x16x32_f16(false, a, false, b,
                                                            (short)0, acc[t], false, false);
        }
    }
}

// ---------------------------------------------------------------------------
// Weight f32 -> f16 converter with zero row-padding
// ---------------------------------------------------------------------------
__global__ void k_f32_to_f16_pad(const float* __restrict__ in, _Float16* __restrict__ out,
                                 int n_valid, int n_total)
{
    int i = blockIdx.x * blockDim.x + threadIdx.x;
    if (i < n_total) out[i] = (i < n_valid) ? (_Float16)in[i] : (_Float16)0.f;
}

// ---------------------------------------------------------------------------
// Axial depthwise convs + residual + biases, transpose NCHW -> token-major f16
// ---------------------------------------------------------------------------
__global__ void k_axial(const float* __restrict__ x,
                        const float* __restrict__ wh, const float* __restrict__ bh,
                        const float* __restrict__ ww, const float* __restrict__ bw,
                        _Float16* __restrict__ xa)
{
    int gid = blockIdx.x * blockDim.x + threadIdx.x;
    if (gid >= BATCH * CDIM * HDIM * WDIM) return;
    int w = gid & 63, h = (gid >> 6) & 63, c = (gid >> 12) & 63, b = gid >> 18;
    const float* xp = x + ((size_t)(b * CDIM + c)) * LSEQ;
    float xc  = xp[h * WDIM + w];
    float xm1 = (h > 0)  ? xp[(h - 1) * WDIM + w] : 0.f;
    float xp1 = (h < 63) ? xp[(h + 1) * WDIM + w] : 0.f;
    float ym1 = (w > 0)  ? xp[h * WDIM + w - 1]   : 0.f;
    float yp1 = (w < 63) ? xp[h * WDIM + w + 1]   : 0.f;
    float vh = wh[c * 3 + 0] * xm1 + wh[c * 3 + 1] * xc + wh[c * 3 + 2] * xp1;
    float vw = ww[c * 3 + 0] * ym1 + ww[c * 3 + 1] * xc + ww[c * 3 + 2] * yp1;
    float v = xc + vh + bh[c] + vw + bw[c];
    int l = h * WDIM + w;
    xa[((size_t)(b * LSEQ + l)) * CDIM + c] = (_Float16)v;
}

// ---------------------------------------------------------------------------
// 1x1 conv (WMMA, NT=4 covers all N=64) + BN(eval) + ReLU -> seq[BL][64] f32
// ---------------------------------------------------------------------------
__global__ void k_gemm_conv_bn(const _Float16* __restrict__ xa, const _Float16* __restrict__ wconv,
                               const float* __restrict__ conv_b,
                               const float* __restrict__ bn_g, const float* __restrict__ bn_b,
                               const float* __restrict__ bn_m, const float* __restrict__ bn_v,
                               float* __restrict__ seq)
{
    int wave  = threadIdx.x >> 5;
    int tileM = blockIdx.x * 8 + wave;       // 0..1023
    v8f acc[4];
    wmma_gemm_multi<CDIM, 4>(xa, CDIM, wconv, CDIM, tileM, 0, acc);
    int lane = threadIdx.x & 31, half = lane >> 4, idx = lane & 15;
#pragma unroll
    for (int t = 0; t < 4; ++t) {
        int n = t * 16 + idx;
        float sc = bn_g[n] * rsqrtf(bn_v[n] + 1e-5f);
        float bs = (conv_b[n] - bn_m[n]) * sc + bn_b[n];
#pragma unroll
        for (int r = 0; r < 8; ++r) {
            int m = tileM * 16 + half * 8 + r;
            float v = acc[t][r] * sc + bs;
            seq[(size_t)m * CDIM + n] = v > 0.f ? v : 0.f;
        }
    }
}

// ---------------------------------------------------------------------------
// LayerNorm over C=64, one wave per token (2 channels per lane) -> hn f16
// ---------------------------------------------------------------------------
__global__ void k_layernorm(const float* __restrict__ seq,
                            const float* __restrict__ ln_g, const float* __restrict__ ln_b,
                            _Float16* __restrict__ hn)
{
    int lane = threadIdx.x & 31;
    int tok  = blockIdx.x * 8 + (threadIdx.x >> 5);
    const float* row = seq + (size_t)tok * CDIM;
    float a = row[lane], b = row[lane + 32];
    float s = a + b;
#pragma unroll
    for (int off = 16; off; off >>= 1) s += __shfl_xor(s, off, 32);
    float mu = s * (1.f / 64.f);
    float da = a - mu, db = b - mu;
    float vs = da * da + db * db;
#pragma unroll
    for (int off = 16; off; off >>= 1) vs += __shfl_xor(vs, off, 32);
    float inv = rsqrtf(vs * (1.f / 64.f) + 1e-5f);
    _Float16* orow = hn + (size_t)tok * CDIM;
    orow[lane]      = (_Float16)(da * inv * ln_g[lane] + ln_b[lane]);
    orow[lane + 32] = (_Float16)(db * inv * ln_g[lane + 32] + ln_b[lane + 32]);
}

// ---------------------------------------------------------------------------
// in_proj WMMA GEMM (NT=4, grid.x=4 covers N=256):
//   [BL,64] x [256,64]^T -> xm f16 [BL,128], z f32 [BL,128]
// ---------------------------------------------------------------------------
__global__ void k_gemm_inproj(const _Float16* __restrict__ hn, const _Float16* __restrict__ wip,
                              _Float16* __restrict__ xm, float* __restrict__ z)
{
    int wave  = threadIdx.x >> 5;
    int tileM = blockIdx.y * 8 + wave;
    int tileNbase = blockIdx.x * 4;          // 0,4,8,12
    v8f acc[4];
    wmma_gemm_multi<CDIM, 4>(hn, CDIM, wip, CDIM, tileM, tileNbase, acc);
    int lane = threadIdx.x & 31, half = lane >> 4, idx = lane & 15;
#pragma unroll
    for (int t = 0; t < 4; ++t) {
        int n = (tileNbase + t) * 16 + idx;
#pragma unroll
        for (int r = 0; r < 8; ++r) {
            int m = tileM * 16 + half * 8 + r;
            if (n < DIN) xm[(size_t)m * DIN + n] = (_Float16)acc[t][r];
            else         z[(size_t)m * DIN + (n - DIN)] = acc[t][r];
        }
    }
}

// ---------------------------------------------------------------------------
// Causal depthwise conv1d (k=4, pad 3 left) + SiLU  -> xc (f16 + f32)
// ---------------------------------------------------------------------------
__global__ void k_dwconv_silu(const _Float16* __restrict__ xm,
                              const float* __restrict__ wc, const float* __restrict__ bc,
                              _Float16* __restrict__ xc16, float* __restrict__ xc32)
{
    int gid = blockIdx.x * blockDim.x + threadIdx.x;
    if (gid >= BL * DIN) return;
    int d = gid & (DIN - 1);
    int row = gid >> 7;                  // b*L + l
    int l = row & (LSEQ - 1);
    float acc = bc[d];
#pragma unroll
    for (int j = 0; j < 4; ++j) {
        int ls = l - 3 + j;
        if (ls >= 0) acc += wc[d * 4 + j] * (float)xm[(size_t)(row - 3 + j) * DIN + d];
    }
    float s = acc / (1.f + __expf(-acc));
    xc16[gid] = (_Float16)s;
    xc32[gid] = s;
}

// ---------------------------------------------------------------------------
// x_proj WMMA GEMM (NT=3, weights padded to 48 rows):
//   [BL,128] x [36,128]^T -> dbc f32 [BL,36]
// ---------------------------------------------------------------------------
__global__ void k_gemm_xproj(const _Float16* __restrict__ xc16, const _Float16* __restrict__ wxp,
                             float* __restrict__ dbc)
{
    int wave  = threadIdx.x >> 5;
    int tileM = blockIdx.x * 8 + wave;
    v8f acc[3];
    wmma_gemm_multi<DIN, 3>(xc16, DIN, wxp, DIN, tileM, 0, acc);
    int lane = threadIdx.x & 31, half = lane >> 4, idx = lane & 15;
#pragma unroll
    for (int t = 0; t < 3; ++t) {
        int n = t * 16 + idx;
        if (n < NPROJ) {
#pragma unroll
            for (int r = 0; r < 8; ++r) {
                int m = tileM * 16 + half * 8 + r;
                dbc[(size_t)m * NPROJ + n] = acc[t][r];
            }
        }
    }
}

// ---------------------------------------------------------------------------
// dt = softplus(dt_small @ dt_proj_w^T + b)   [BL,128]
// ---------------------------------------------------------------------------
__global__ void k_dt(const float* __restrict__ dbc, const float* __restrict__ dtw,
                     const float* __restrict__ dtb, float* __restrict__ dt)
{
    int gid = blockIdx.x * blockDim.x + threadIdx.x;
    if (gid >= BL * DIN) return;
    int d = gid & (DIN - 1);
    int row = gid >> 7;
    const float* p = dbc + (size_t)row * NPROJ;
    float v = dtb[d];
#pragma unroll
    for (int r = 0; r < DTR; ++r) v += p[r] * dtw[d * DTR + r];
    dt[gid] = (v > 20.f) ? v : log1pf(__expf(v));
}

// ---------------------------------------------------------------------------
// Selective scan. 8 blocks (batch x half-of-D), 1024 threads; thread owns one
// (d,s) state; 16-lane shfl_xor reduce for y_d = sum_s h*C.  Prefetch the row
// 4 steps ahead (global_prefetch_b8) to hide load latency behind the serial
// exp+FMA dependency chain.
// ---------------------------------------------------------------------------
__global__ void __launch_bounds__(1024)
k_scan(const float* __restrict__ dt, const float* __restrict__ xc32,
       const float* __restrict__ dbc, const float* __restrict__ A_log,
       float* __restrict__ ys)
{
    int b = blockIdx.x >> 1;
    int halfD = blockIdx.x & 1;
    int s = threadIdx.x & 15;
    int d = (threadIdx.x >> 4) + 64 * halfD;
    float A = -__expf(A_log[d * DST + s]);
    float h = 0.f;
    size_t base = (size_t)b * LSEQ;
    for (int l = 0; l < LSEQ; ++l) {
        size_t row = base + l;
        if (l + 4 < LSEQ) {
            size_t prow = base + l + 4;
            __builtin_prefetch(dt   + prow * DIN + d, 0, 0);
            __builtin_prefetch(xc32 + prow * DIN + d, 0, 0);
            __builtin_prefetch(dbc  + prow * NPROJ + DTR + s, 0, 0);
        }
        float dtv = dt[row * DIN + d];
        float xcv = xc32[row * DIN + d];
        float Bv  = dbc[row * NPROJ + DTR + s];
        float Cv  = dbc[row * NPROJ + DTR + DST + s];
        h = h * __expf(dtv * A) + dtv * Bv * xcv;
        float p = h * Cv;
        p += __shfl_xor(p, 1, 32);
        p += __shfl_xor(p, 2, 32);
        p += __shfl_xor(p, 4, 32);
        p += __shfl_xor(p, 8, 32);
        if (s == 0) ys[row * DIN + d] = p;
    }
}

// ---------------------------------------------------------------------------
// Gate: yact = (ys + xc*Dp) * silu(z)  -> f16 for out_proj GEMM
// ---------------------------------------------------------------------------
__global__ void k_gate(const float* __restrict__ ys, const float* __restrict__ xc32,
                       const float* __restrict__ Dp, const float* __restrict__ z,
                       _Float16* __restrict__ yact)
{
    int gid = blockIdx.x * blockDim.x + threadIdx.x;
    if (gid >= BL * DIN) return;
    int d = gid & (DIN - 1);
    float v = ys[gid] + xc32[gid] * Dp[d];
    float zz = z[gid];
    v *= zz / (1.f + __expf(-zz));
    yact[gid] = (_Float16)v;
}

// ---------------------------------------------------------------------------
// out_proj WMMA GEMM (NT=4 covers N=64):
//   [BL,128] x [64,128]^T + residual(seq) + NCHW transpose -> d_out
// ---------------------------------------------------------------------------
__global__ void k_gemm_out(const _Float16* __restrict__ yact, const _Float16* __restrict__ wout,
                           const float* __restrict__ seq, float* __restrict__ out)
{
    int wave  = threadIdx.x >> 5;
    int tileM = blockIdx.x * 8 + wave;
    v8f acc[4];
    wmma_gemm_multi<DIN, 4>(yact, DIN, wout, DIN, tileM, 0, acc);
    int lane = threadIdx.x & 31, half = lane >> 4, idx = lane & 15;
#pragma unroll
    for (int t = 0; t < 4; ++t) {
        int n = t * 16 + idx;
#pragma unroll
        for (int r = 0; r < 8; ++r) {
            int m = tileM * 16 + half * 8 + r;
            int bidx = m >> 12, l = m & (LSEQ - 1);
            float v = acc[t][r] + seq[(size_t)m * CDIM + n];
            out[((size_t)(bidx * CDIM + n)) * LSEQ + l] = v;
        }
    }
}

// ---------------------------------------------------------------------------
extern "C" void kernel_launch(void* const* d_in, const int* in_sizes, int n_in,
                              void* d_out, int out_size, void* d_ws, size_t ws_size,
                              hipStream_t stream)
{
    const float* x        = (const float*)d_in[0];
    const float* dwh_w    = (const float*)d_in[1];
    const float* dwh_b    = (const float*)d_in[2];
    const float* dww_w    = (const float*)d_in[3];
    const float* dww_b    = (const float*)d_in[4];
    const float* conv_w   = (const float*)d_in[5];
    const float* conv_b   = (const float*)d_in[6];
    const float* bn_g     = (const float*)d_in[7];
    const float* bn_b     = (const float*)d_in[8];
    const float* bn_m     = (const float*)d_in[9];
    const float* bn_v     = (const float*)d_in[10];
    const float* ln_g     = (const float*)d_in[11];
    const float* ln_b     = (const float*)d_in[12];
    const float* in_proj_w  = (const float*)d_in[13];
    const float* convd_w    = (const float*)d_in[14];
    const float* convd_b    = (const float*)d_in[15];
    const float* x_proj_w   = (const float*)d_in[16];
    const float* dt_proj_w  = (const float*)d_in[17];
    const float* dt_proj_b  = (const float*)d_in[18];
    const float* A_log      = (const float*)d_in[19];
    const float* Dp         = (const float*)d_in[20];
    const float* out_proj_w = (const float*)d_in[21];
    float* out = (float*)d_out;

    // Bump-allocate workspace (256B aligned)
    char* p = (char*)d_ws;
    auto alloc = [&](size_t bytes) -> void* {
        void* r = (void*)p;
        p += (bytes + 255) & ~(size_t)255;
        return r;
    };
    _Float16* w_conv_h = (_Float16*)alloc((size_t)CDIM * CDIM * 2);
    _Float16* w_ip_h   = (_Float16*)alloc((size_t)2 * DIN * CDIM * 2);
    _Float16* w_xp_h   = (_Float16*)alloc((size_t)NPROJ_PAD * DIN * 2);
    _Float16* w_op_h   = (_Float16*)alloc((size_t)CDIM * DIN * 2);
    _Float16* xa       = (_Float16*)alloc((size_t)BL * CDIM * 2);
    float*    seq      = (float*)   alloc((size_t)BL * CDIM * 4);
    _Float16* hn       = (_Float16*)alloc((size_t)BL * CDIM * 2);
    _Float16* xm       = (_Float16*)alloc((size_t)BL * DIN * 2);
    float*    zbuf     = (float*)   alloc((size_t)BL * DIN * 4);
    _Float16* xc16     = (_Float16*)alloc((size_t)BL * DIN * 2);
    float*    xc32     = (float*)   alloc((size_t)BL * DIN * 4);
    float*    dbc      = (float*)   alloc((size_t)BL * NPROJ * 4);
    float*    dtb      = (float*)   alloc((size_t)BL * DIN * 4);
    float*    ysb      = (float*)   alloc((size_t)BL * DIN * 4);
    _Float16* yact     = (_Float16*)alloc((size_t)BL * DIN * 2);

    // Weight conversions to f16 (x_proj zero-padded 36 -> 48 rows)
    k_f32_to_f16_pad<<<(CDIM * CDIM + 255) / 256, 256, 0, stream>>>(
        conv_w, w_conv_h, CDIM * CDIM, CDIM * CDIM);
    k_f32_to_f16_pad<<<(2 * DIN * CDIM + 255) / 256, 256, 0, stream>>>(
        in_proj_w, w_ip_h, 2 * DIN * CDIM, 2 * DIN * CDIM);
    k_f32_to_f16_pad<<<(NPROJ_PAD * DIN + 255) / 256, 256, 0, stream>>>(
        x_proj_w, w_xp_h, NPROJ * DIN, NPROJ_PAD * DIN);
    k_f32_to_f16_pad<<<(CDIM * DIN + 255) / 256, 256, 0, stream>>>(
        out_proj_w, w_op_h, CDIM * DIN, CDIM * DIN);

    // 1. Axial depthwise + residual -> token-major f16
    k_axial<<<(BATCH * CDIM * HDIM * WDIM + 255) / 256, 256, 0, stream>>>(
        x, dwh_w, dwh_b, dww_w, dww_b, xa);

    // 2. 1x1 conv (WMMA) + BN + ReLU -> seq
    k_gemm_conv_bn<<<BL / 16 / 8, 256, 0, stream>>>(
        xa, w_conv_h, conv_b, bn_g, bn_b, bn_m, bn_v, seq);

    // 3. LayerNorm -> hn f16
    k_layernorm<<<BL / 8, 256, 0, stream>>>(seq, ln_g, ln_b, hn);

    // 4. in_proj (WMMA) -> xm f16, z f32
    k_gemm_inproj<<<dim3(4, BL / 16 / 8), 256, 0, stream>>>(hn, w_ip_h, xm, zbuf);

    // 5. causal depthwise conv1d + SiLU
    k_dwconv_silu<<<(BL * DIN + 255) / 256, 256, 0, stream>>>(xm, convd_w, convd_b, xc16, xc32);

    // 6. x_proj (WMMA, N padded to 48) -> dbc
    k_gemm_xproj<<<BL / 16 / 8, 256, 0, stream>>>(xc16, w_xp_h, dbc);

    // 7. dt_proj + softplus
    k_dt<<<(BL * DIN + 255) / 256, 256, 0, stream>>>(dbc, dt_proj_w, dt_proj_b, dtb);

    // 8. selective scan (with lookahead prefetch)
    k_scan<<<BATCH * 2, 1024, 0, stream>>>(dtb, xc32, dbc, A_log, ysb);

    // 9. gate
    k_gate<<<(BL * DIN + 255) / 256, 256, 0, stream>>>(ysb, xc32, Dp, zbuf, yact);

    // 10. out_proj (WMMA) + residual + NCHW transpose -> d_out
    k_gemm_out<<<BL / 16 / 8, 256, 0, stream>>>(yact, w_op_h, seq, out);
}
